// BilstmCrf_9835475108292
// MI455X (gfx1250) — compile-verified, matrix operand order checked by reading, not run on hardware
//
#include <hip/hip_runtime.h>
#include <hip/hip_bf16.h>
#include <math.h>

typedef __attribute__((ext_vector_type(16))) _Float16 v16h;
typedef __attribute__((ext_vector_type(8)))  float    v8f;

#define B_     128
#define T_     256
#define H_     512
#define G4_    2048
#define KC_    62
#define START_ 60
#define STOP_  61
#define NBLK_REC 128

union HFrag { v16h v; int4 q[2]; };

__device__ __forceinline__ float sigm(float x) { return 1.0f / (1.0f + expf(-x)); }

// ---------------- elementwise prep kernels ----------------

__global__ void __launch_bounds__(256) cast_f32_f16(const float* __restrict__ s,
                                                    _Float16* __restrict__ d, int n) {
  int i = blockIdx.x * 256 + threadIdx.x;
  if (i < n) d[i] = (_Float16)s[i];
}

// pad w_tag (62x1024) -> (64x1024) f16, b_tag (62) -> (64) f32
__global__ void __launch_bounds__(256) pad_wtag(const float* __restrict__ wt,
                                                const float* __restrict__ bt,
                                                _Float16* __restrict__ wth,
                                                float* __restrict__ bth) {
  int i = blockIdx.x * 256 + threadIdx.x;     // 65536 total
  int r = i >> 10, k = i & 1023;
  wth[i] = (_Float16)((r < KC_) ? wt[r * 1024 + k] : 0.0f);
  if (i < 64) bth[i] = (i < KC_) ? bt[i] : 0.0f;
}

// build x0 f16 (B*T, 256) = concat(char_emb[bc], intent_emb[bi])
__global__ void __launch_bounds__(256) embed_concat(const int* __restrict__ bc,
                                                    const int* __restrict__ bi,
                                                    const float* __restrict__ ce,
                                                    const float* __restrict__ ie,
                                                    _Float16* __restrict__ x) {
  int gid = blockIdx.x * 256 + threadIdx.x;   // B*T*256
  int row = gid >> 8;
  int d   = gid & 255;
  int b   = row >> 8;                          // T_ == 256
  float v;
  if (d < 200) v = ce[(size_t)bc[row] * 200 + d];
  else         v = ie[(size_t)bi[b] * 56 + (d - 200)];
  x[gid] = (_Float16)v;
}

// -------- generic WMMA GEMM: C = A(MxK,f16) * W(NxK,f16)^T + bias ---------
// Register-blocked: each wave computes a 32x64 macro-tile (2 M-tiles x 4 N-tiles).
// Requires M % 32 == 0, N % 64 == 0, K % 32 == 0.

__global__ void __launch_bounds__(256) gemm_wmma(const _Float16* __restrict__ A,
                                                 const _Float16* __restrict__ W,
                                                 const float* __restrict__ bias,
                                                 void* __restrict__ C,
                                                 int M, int N, int K, int c_f16) {
  int wave = threadIdx.x >> 5;
  int lane = threadIdx.x & 31;
  int wtile = blockIdx.x * 8 + wave;
  int nsup = N >> 6;                 // 64-wide super-tiles
  int ms = wtile / nsup, ns = wtile % nsup;
  if (ms >= (M >> 5)) return;
  int r15 = lane & 15, hi = lane >> 4;
  int mbase = ms * 32, nbase = ns * 64;

  const _Float16* arow0 = A + (size_t)(mbase + r15) * K;
  const _Float16* arow1 = A + (size_t)(mbase + 16 + r15) * K;
  const _Float16* wrow0 = W + (size_t)(nbase + r15) * K;
  const _Float16* wrow1 = W + (size_t)(nbase + 16 + r15) * K;
  const _Float16* wrow2 = W + (size_t)(nbase + 32 + r15) * K;
  const _Float16* wrow3 = W + (size_t)(nbase + 48 + r15) * K;

  v8f acc[2][4];
  #pragma unroll
  for (int qm = 0; qm < 2; ++qm)
    #pragma unroll
    for (int qn = 0; qn < 4; ++qn)
      acc[qm][qn] = (v8f){};

  #pragma unroll 2
  for (int kb = 0; kb < K; kb += 32) {
    int ko = kb + hi * 8;
    HFrag fa0, fa1, fb0, fb1, fb2, fb3;
    fa0.q[0] = *reinterpret_cast<const int4*>(arow0 + ko);
    fa0.q[1] = *reinterpret_cast<const int4*>(arow0 + ko + 16);
    fa1.q[0] = *reinterpret_cast<const int4*>(arow1 + ko);
    fa1.q[1] = *reinterpret_cast<const int4*>(arow1 + ko + 16);
    fb0.q[0] = *reinterpret_cast<const int4*>(wrow0 + ko);
    fb0.q[1] = *reinterpret_cast<const int4*>(wrow0 + ko + 16);
    fb1.q[0] = *reinterpret_cast<const int4*>(wrow1 + ko);
    fb1.q[1] = *reinterpret_cast<const int4*>(wrow1 + ko + 16);
    fb2.q[0] = *reinterpret_cast<const int4*>(wrow2 + ko);
    fb2.q[1] = *reinterpret_cast<const int4*>(wrow2 + ko + 16);
    fb3.q[0] = *reinterpret_cast<const int4*>(wrow3 + ko);
    fb3.q[1] = *reinterpret_cast<const int4*>(wrow3 + ko + 16);
    acc[0][0] = __builtin_amdgcn_wmma_f32_16x16x32_f16(false, fa0.v, false, fb0.v, (short)0, acc[0][0], false, false);
    acc[0][1] = __builtin_amdgcn_wmma_f32_16x16x32_f16(false, fa0.v, false, fb1.v, (short)0, acc[0][1], false, false);
    acc[0][2] = __builtin_amdgcn_wmma_f32_16x16x32_f16(false, fa0.v, false, fb2.v, (short)0, acc[0][2], false, false);
    acc[0][3] = __builtin_amdgcn_wmma_f32_16x16x32_f16(false, fa0.v, false, fb3.v, (short)0, acc[0][3], false, false);
    acc[1][0] = __builtin_amdgcn_wmma_f32_16x16x32_f16(false, fa1.v, false, fb0.v, (short)0, acc[1][0], false, false);
    acc[1][1] = __builtin_amdgcn_wmma_f32_16x16x32_f16(false, fa1.v, false, fb1.v, (short)0, acc[1][1], false, false);
    acc[1][2] = __builtin_amdgcn_wmma_f32_16x16x32_f16(false, fa1.v, false, fb2.v, (short)0, acc[1][2], false, false);
    acc[1][3] = __builtin_amdgcn_wmma_f32_16x16x32_f16(false, fa1.v, false, fb3.v, (short)0, acc[1][3], false, false);
  }

  #pragma unroll
  for (int qn = 0; qn < 4; ++qn) {
    int n = nbase + qn * 16 + r15;
    float bv = bias ? bias[n] : 0.0f;
    #pragma unroll
    for (int qm = 0; qm < 2; ++qm) {
      #pragma unroll
      for (int r = 0; r < 8; ++r) {
        int m = mbase + qm * 16 + r + hi * 8;
        float val = acc[qm][qn][r] + bv;
        if (c_f16) ((_Float16*)C)[(size_t)m * N + n] = (_Float16)val;
        else       ((float*)C)[(size_t)m * N + n] = val;
      }
    }
  }
}

// ---------------- persistent recurrent LSTM kernel ----------------

__device__ __forceinline__ void grid_sync(unsigned* bar, unsigned target) {
  __syncthreads();
  if (threadIdx.x == 0) {
    __threadfence();
    atomicAdd(bar, 1u);
    while (__hip_atomic_load(bar, __ATOMIC_RELAXED, __HIP_MEMORY_SCOPE_AGENT) < target) {
      __builtin_amdgcn_s_sleep(1);
    }
  }
  __syncthreads();
  __threadfence();
}

__global__ void __launch_bounds__(256) lstm_step_kernel(
    const _Float16* __restrict__ G,    // (B*T) x 2048 precomputed x*Wih^T + b (f16)
    const _Float16* __restrict__ whh,  // 2048 x 512 (f16)
    _Float16* __restrict__ hs,         // (B*T) x 1024 output (write at dirOff)
    _Float16* __restrict__ h_buf,      // B x 512 f16 (pre-zeroed)
    float* __restrict__ c_buf,         // B x 512 f32 (pre-zeroed)
    float* __restrict__ gates,         // B x 2048 f32 scratch
    unsigned* __restrict__ bar,        // pre-zeroed counter
    int reverse, int dirOff) {
  int wave = threadIdx.x >> 5, lane = threadIdx.x & 31;
  int wtile = blockIdx.x * 8 + wave;   // 1024 waves == 8x128 tiles of (128 x 2048)
  int mt = wtile >> 7, nt = wtile & 127;
  int r15 = lane & 15, hi = lane >> 4;
  const _Float16* arow = h_buf + (size_t)(mt * 16 + r15) * H_;
  const _Float16* wrow = whh + (size_t)(nt * 16 + r15) * H_;
  int n = nt * 16 + r15;
  int gid = blockIdx.x * 256 + threadIdx.x;

  // whh tile for this wave is invariant over time: keep it register-resident
  // (16 K-fragments x 8 VGPRs = 128 VGPRs) for the whole serial recurrence.
  HFrag fbw[16];
  #pragma unroll
  for (int ks = 0; ks < 16; ++ks) {
    int ko = ks * 32 + hi * 8;
    fbw[ks].q[0] = *reinterpret_cast<const int4*>(wrow + ko);
    fbw[ks].q[1] = *reinterpret_cast<const int4*>(wrow + ko + 16);
  }

  unsigned phase = 0;
  for (int t = 0; t < T_; ++t) {
    int tt = reverse ? (T_ - 1 - t) : t;
    // phase A: gates = h_prev @ whh^T + G[:, tt, :]
    v8f acc = {};
    #pragma unroll
    for (int ks = 0; ks < 16; ++ks) {
      HFrag fa;
      int ko = ks * 32 + hi * 8;
      fa.q[0] = *reinterpret_cast<const int4*>(arow + ko);
      fa.q[1] = *reinterpret_cast<const int4*>(arow + ko + 16);
      acc = __builtin_amdgcn_wmma_f32_16x16x32_f16(false, fa.v, false, fbw[ks].v,
                                                   (short)0, acc, false, false);
    }
    #pragma unroll
    for (int r = 0; r < 8; ++r) {
      int m = mt * 16 + r + hi * 8;
      gates[(size_t)m * G4_ + n] = acc[r] + (float)G[((size_t)m * T_ + tt) * G4_ + n];
    }
    // warm L2/WGP$ with next timestep's G slice while we sit in the barrier
    // and phase B (G streams from HBM; address is known one step ahead).
    if (t + 1 < T_) {
      int tn = reverse ? (T_ - 2 - t) : (t + 1);
      #pragma unroll
      for (int r = 0; r < 8; ++r) {
        int m = mt * 16 + r + hi * 8;
        __builtin_prefetch(&G[((size_t)m * T_ + tn) * G4_ + n], 0, 3);
      }
    }
    grid_sync(bar, (++phase) * NBLK_REC);
    // phase B: pointwise cell update (65536 elems over 32768 threads)
    for (int e = gid; e < B_ * H_; e += NBLK_REC * 256) {
      int b = e >> 9, hn = e & 511;
      const float* g = gates + (size_t)b * G4_;
      float gi = g[hn], gf = g[H_ + hn], gg = g[2 * H_ + hn], go = g[3 * H_ + hn];
      float c = sigm(gf) * c_buf[e] + sigm(gi) * tanhf(gg);
      float h = sigm(go) * tanhf(c);
      c_buf[e] = c;
      _Float16 hh = (_Float16)h;
      h_buf[e] = hh;
      hs[((size_t)b * T_ + tt) * 1024 + dirOff + hn] = hh;
    }
    grid_sync(bar, (++phase) * NBLK_REC);
  }
}

// ---------------- CRF kernels (fp32) ----------------

__global__ void __launch_bounds__(64) crf_forward_kernel(const float* __restrict__ feats,
                                                         const int* __restrict__ lens,
                                                         const float* __restrict__ trans,
                                                         float* __restrict__ logZ) {
  __shared__ float sa[64];
  int b = blockIdx.x, j = threadIdx.x;
  int len = lens[b];
  const float* fb = feats + (size_t)b * T_ * 64;
  float alpha = -1e30f;
  if (j < KC_) alpha = trans[START_ * KC_ + j] + fb[j];
  for (int t = 1; t < T_; ++t) {
    sa[j] = alpha;
    __syncthreads();
    if (t < len && j < KC_) {
      float m = -1e30f;
      for (int i = 0; i < KC_; ++i) m = fmaxf(m, sa[i] + trans[i * KC_ + j]);
      float s = 0.0f;
      for (int i = 0; i < KC_; ++i) s += expf(sa[i] + trans[i * KC_ + j] - m);
      alpha = m + logf(s) + fb[t * 64 + j];
    }
    __syncthreads();
  }
  sa[j] = (j < KC_) ? alpha + trans[j * KC_ + STOP_] : -1e30f;
  __syncthreads();
  if (j == 0) {
    float m = -1e30f;
    for (int i = 0; i < KC_; ++i) m = fmaxf(m, sa[i]);
    float s = 0.0f;
    for (int i = 0; i < KC_; ++i) s += expf(sa[i] - m);
    logZ[b] = m + logf(s);
  }
}

__global__ void __launch_bounds__(128) crf_gold_kernel(const float* __restrict__ feats,
                                                       const int* __restrict__ lens,
                                                       const int* __restrict__ labels,
                                                       const float* __restrict__ trans,
                                                       float* __restrict__ gold) {
  int b = blockIdx.x * blockDim.x + threadIdx.x;
  if (b >= B_) return;
  int len = lens[b];
  const float* fb = feats + (size_t)b * T_ * 64;
  const int* lb = labels + b * T_;
  float s = 0.0f;
  for (int t = 0; t < T_; ++t) if (t < len) s += fb[t * 64 + lb[t]];
  for (int t = 1; t < T_; ++t) if (t < len) s += trans[lb[t - 1] * KC_ + lb[t]];
  s += trans[START_ * KC_ + lb[0]];
  s += trans[lb[len - 1] * KC_ + STOP_];
  gold[b] = s;
}

__global__ void __launch_bounds__(64) viterbi_kernel(const float* __restrict__ feats,
                                                     const int* __restrict__ lens,
                                                     const float* __restrict__ trans,
                                                     unsigned char* __restrict__ bp,
                                                     float* __restrict__ out_tags) {
  __shared__ float sa[64];
  int b = blockIdx.x, j = threadIdx.x;
  int len = lens[b];
  const float* fb = feats + (size_t)b * T_ * 64;
  float alpha = -1e30f;
  if (j < KC_) alpha = trans[START_ * KC_ + j] + fb[j];
  for (int t = 1; t < T_; ++t) {
    sa[j] = alpha;
    __syncthreads();
    unsigned char bj = (unsigned char)j;
    if (j < KC_ && t < len) {
      float best = sa[0] + trans[0 * KC_ + j];
      int bidx = 0;
      for (int i = 1; i < KC_; ++i) {
        float v = sa[i] + trans[i * KC_ + j];
        if (v > best) { best = v; bidx = i; }
      }
      alpha = best + fb[t * 64 + j];
      bj = (unsigned char)bidx;
    }
    bp[((size_t)b * T_ + t) * 64 + j] = bj;
    __syncthreads();
  }
  sa[j] = (j < KC_) ? alpha + trans[j * KC_ + STOP_] : -1e30f;
  __syncthreads();
  if (j == 0) {
    int tag = 0;
    float best = sa[0];
    for (int i = 1; i < KC_; ++i) if (sa[i] > best) { best = sa[i]; tag = i; }
    out_tags[b * T_ + T_ - 1] = (T_ - 1 < len) ? (float)tag : 0.0f;
    for (int t = T_ - 1; t >= 1; --t) {
      tag = bp[((size_t)b * T_ + t) * 64 + tag];
      out_tags[b * T_ + t - 1] = (t - 1 < len) ? (float)tag : 0.0f;
    }
  }
}

__global__ void __launch_bounds__(128) loss_kernel(const float* __restrict__ logZ,
                                                   const float* __restrict__ gold,
                                                   float* __restrict__ out) {
  __shared__ float s[128];
  int i = threadIdx.x;
  s[i] = logZ[i] - gold[i];
  __syncthreads();
  for (int st = 64; st > 0; st >>= 1) {
    if (i < st) s[i] += s[i + st];
    __syncthreads();
  }
  if (i == 0) out[0] = s[0];
}

// ---------------- host launcher ----------------

extern "C" void kernel_launch(void* const* d_in, const int* in_sizes, int n_in,
                              void* d_out, int out_size, void* d_ws, size_t ws_size,
                              hipStream_t stream) {
  (void)in_sizes; (void)n_in; (void)out_size; (void)ws_size;
  const int*   bc     = (const int*)d_in[0];
  const int*   bi     = (const int*)d_in[1];
  const int*   lens   = (const int*)d_in[2];
  const int*   labels = (const int*)d_in[4];
  const float* ce     = (const float*)d_in[5];
  const float* ie     = (const float*)d_in[6];
  const float* wih0   = (const float*)d_in[7];
  const float* whh0   = (const float*)d_in[8];
  const float* b0     = (const float*)d_in[9];
  const float* wih1   = (const float*)d_in[10];
  const float* whh1   = (const float*)d_in[11];
  const float* b1     = (const float*)d_in[12];
  const float* wtag   = (const float*)d_in[13];
  const float* btag   = (const float*)d_in[14];
  const float* trans  = (const float*)d_in[15];

  char* ws = (char*)d_ws;
  size_t off = 0;
  auto alloc = [&](size_t bytes) -> char* {
    char* p = ws + off;
    off += (bytes + 255) & ~(size_t)255;
    return p;
  };
  unsigned* ctrl    = (unsigned*)alloc(1024);                 // 4 barrier counters (64B apart)
  float* logZ       = (float*)alloc(B_ * 4);
  float* gold       = (float*)alloc(B_ * 4);
  float* btag_p     = (float*)alloc(64 * 4);
  _Float16* h_buf   = (_Float16*)alloc((size_t)B_ * H_ * 2);
  float* c_buf      = (float*)alloc((size_t)B_ * H_ * 4);
  float* gates      = (float*)alloc((size_t)B_ * G4_ * 4);
  _Float16* x0      = (_Float16*)alloc((size_t)B_ * T_ * 256 * 2);
  _Float16* wih0h   = (_Float16*)alloc((size_t)2 * G4_ * 256 * 2);
  _Float16* whh0h   = (_Float16*)alloc((size_t)2 * G4_ * H_ * 2);
  _Float16* wih1h   = (_Float16*)alloc((size_t)2 * G4_ * 1024 * 2);
  _Float16* whh1h   = (_Float16*)alloc((size_t)2 * G4_ * H_ * 2);
  _Float16* wtagh   = (_Float16*)alloc((size_t)64 * 1024 * 2);
  _Float16* G       = (_Float16*)alloc((size_t)B_ * T_ * G4_ * 2);
  _Float16* hs0     = (_Float16*)alloc((size_t)B_ * T_ * 1024 * 2);
  _Float16* hs1     = (_Float16*)alloc((size_t)B_ * T_ * 1024 * 2);
  float* feats      = (float*)alloc((size_t)B_ * T_ * 64 * 4);
  unsigned char* bp = (unsigned char*)alloc((size_t)B_ * T_ * 64);

  const int M = B_ * T_;  // 32768

  hipMemsetAsync(ctrl, 0, 1024, stream);

  // weight casts
  int n0 = 2 * G4_ * 256;
  cast_f32_f16<<<(n0 + 255) / 256, 256, 0, stream>>>(wih0, wih0h, n0);
  int n1 = 2 * G4_ * H_;
  cast_f32_f16<<<(n1 + 255) / 256, 256, 0, stream>>>(whh0, whh0h, n1);
  int n2 = 2 * G4_ * 1024;
  cast_f32_f16<<<(n2 + 255) / 256, 256, 0, stream>>>(wih1, wih1h, n2);
  cast_f32_f16<<<(n1 + 255) / 256, 256, 0, stream>>>(whh1, whh1h, n1);
  pad_wtag<<<256, 256, 0, stream>>>(wtag, btag, wtagh, btag_p);

  // embedding concat
  embed_concat<<<(M * 256) / 256, 256, 0, stream>>>(bc, bi, ce, ie, x0);

  // layer 0
  for (int dir = 0; dir < 2; ++dir) {
    gemm_wmma<<<(M / 32) * (G4_ / 64) / 8, 256, 0, stream>>>(
        x0, wih0h + (size_t)dir * G4_ * 256, b0 + dir * G4_, G, M, G4_, 256, 1);
    hipMemsetAsync(h_buf, 0, (size_t)B_ * H_ * 2, stream);
    hipMemsetAsync(c_buf, 0, (size_t)B_ * H_ * 4, stream);
    lstm_step_kernel<<<NBLK_REC, 256, 0, stream>>>(
        G, whh0h + (size_t)dir * G4_ * H_, hs0, h_buf, c_buf, gates,
        ctrl + dir * 16, dir, dir * H_);
  }
  // layer 1
  for (int dir = 0; dir < 2; ++dir) {
    gemm_wmma<<<(M / 32) * (G4_ / 64) / 8, 256, 0, stream>>>(
        hs0, wih1h + (size_t)dir * G4_ * 1024, b1 + dir * G4_, G, M, G4_, 1024, 1);
    hipMemsetAsync(h_buf, 0, (size_t)B_ * H_ * 2, stream);
    hipMemsetAsync(c_buf, 0, (size_t)B_ * H_ * 4, stream);
    lstm_step_kernel<<<NBLK_REC, 256, 0, stream>>>(
        G, whh1h + (size_t)dir * G4_ * H_, hs1, h_buf, c_buf, gates,
        ctrl + (2 + dir) * 16, dir, dir * H_);
  }

  // feats = hs1 @ w_tag^T + b_tag  (N padded to 64, fp32 out)
  gemm_wmma<<<(M / 32) * (64 / 64) / 8, 256, 0, stream>>>(
      hs1, wtagh, btag_p, feats, M, 64, 1024, 0);

  // CRF
  crf_forward_kernel<<<B_, 64, 0, stream>>>(feats, lens, trans, logZ);
  crf_gold_kernel<<<1, 128, 0, stream>>>(feats, lens, labels, trans, gold);
  viterbi_kernel<<<B_, 64, 0, stream>>>(feats, lens, trans, bp, (float*)d_out + 1);
  loss_kernel<<<1, 128, 0, stream>>>(logZ, gold, (float*)d_out);
}
